// HiggsMultiLinear_80504866996441
// MI455X (gfx1250) — matrix operand order repaired and checked by reading, not compiled, not run on hardware
//
#include <hip/hip_runtime.h>

typedef __attribute__((ext_vector_type(16))) __bf16 v16bf;
typedef __attribute__((ext_vector_type(8)))  __bf16 v8bf;
typedef __attribute__((ext_vector_type(8)))  float  v8f;

#define IN_F   4096
#define OUT_F  4096
#define MROWS  4096            // BATCH * SEQ
#define KCODES (IN_F / 2)

#define MT   256               // M tile per workgroup
#define NT   128               // N tile per workgroup
#define KT   32                // K step (bf16 WMMA depth)
#define LSTR 40                // LDS row stride in halves (32 + 8 pad, keeps 16B align)

// float -> packed 2x bf16 via native conversion (RNE)
__device__ __forceinline__ unsigned pk2(float a, float b) {
  unsigned short ua = __builtin_bit_cast(unsigned short, (__bf16)a);
  unsigned short ub = __builtin_bit_cast(unsigned short, (__bf16)b);
  return (unsigned)ua | ((unsigned)ub << 16);
}

__global__ __launch_bounds__(256)
void higgs_wmma_kernel(const float* __restrict__ x,
                       const int*   __restrict__ codes,
                       const float* __restrict__ grid,
                       const float* __restrict__ scales,
                       const float* __restrict__ bias,
                       float*       __restrict__ out)
{
  __shared__ unsigned short As[MT * LSTR];   // 20 KB  x tile (bf16)
  __shared__ unsigned short Bs[NT * LSTR];   // 10 KB  dequantized (unscaled) W tile
  __shared__ unsigned gridPk[256];           // 1 KB   codebook pre-packed to 2x bf16

  const int t   = threadIdx.x;
  const int wgN = blockIdx.x;                // N tile index (32)
  const int wgM = blockIdx.y;                // M tile index (16)

  // one-time: pack codebook entries (2 floats) into one dword of 2x bf16
  {
    float2 g = ((const float2*)grid)[t & 255];
    gridPk[t & 255] = pk2(g.x, g.y);
  }

  const int lane = t & 31;
  const int wave = t >> 5;                   // 0..7
  const int wm   = wave & 3;                 // 4 waves along M  (64 rows each)
  const int wn   = wave >> 2;                // 2 waves along N  (64 cols each)
  const int col  = lane & 15;
  const int hi16 = lane >> 4;                // 0/1 (half-wave select)

  // B-dequant mapping: 2 threads per output row, 8 codes (16 K values) each
  const int  bn    = t >> 1;                 // 0..127 row within N tile
  const int  bh    = t & 1;                  // which half of the 32-K slab
  const int* cbase = codes + (size_t)(wgN * NT + bn) * KCODES + bh * 8;

  v8f acc[4][4] = {};

  for (int k0 = 0; k0 < IN_F; k0 += KT) {
    // ---- global loads into registers (overlap with previous compute) ----
    float4 av[8];
    int arow[8], acol[8];
#pragma unroll
    for (int i = 0; i < 8; ++i) {
      int slot = i * 256 + t;                // 2048 float4 slots in the 256x32 A tile
      arow[i] = slot >> 3;                   // 8 float4 per 32-wide row
      acol[i] = (slot & 7) << 2;
      av[i] = *(const float4*)(x + (size_t)(wgM * MT + arow[i]) * IN_F + k0 + acol[i]);
    }
    int4 cv0 = *(const int4*)(cbase + (k0 >> 1));
    int4 cv1 = *(const int4*)(cbase + (k0 >> 1) + 4);

    if (k0 + KT < IN_F) {                    // prefetch next K slab
      __builtin_prefetch(x + (size_t)(wgM * MT + arow[0]) * IN_F + (k0 + KT) + acol[0], 0, 0);
      __builtin_prefetch(cbase + ((k0 + KT) >> 1), 0, 0);
    }

    __syncthreads();                         // previous iteration's WMMAs done reading LDS

    // ---- A tile: f32 -> bf16 into LDS ----
#pragma unroll
    for (int i = 0; i < 8; ++i) {
      unsigned* p = (unsigned*)&As[arow[i] * LSTR + acol[i]];
      p[0] = pk2(av[i].x, av[i].y);
      p[1] = pk2(av[i].z, av[i].w);
    }
    // ---- B tile: pure gather from packed codebook (no VALU dequant) ----
    {
      const int cc[8] = {cv0.x, cv0.y, cv0.z, cv0.w, cv1.x, cv1.y, cv1.z, cv1.w};
      uint4 w0, w1;
      w0.x = gridPk[cc[0]]; w0.y = gridPk[cc[1]]; w0.z = gridPk[cc[2]]; w0.w = gridPk[cc[3]];
      w1.x = gridPk[cc[4]]; w1.y = gridPk[cc[5]]; w1.z = gridPk[cc[6]]; w1.w = gridPk[cc[7]];
      uint4* p = (uint4*)&Bs[bn * LSTR + bh * 16];
      p[0] = w0;
      p[1] = w1;
    }

    __syncthreads();                         // tiles visible to all waves

    // ---- fragment loads (ISA 7.12.2 bf16 layouts) ----
    v16bf a[4], b[4];
#pragma unroll
    for (int mi = 0; mi < 4; ++mi) {
      int r  = wm * 64 + mi * 16 + col;
      int kb = hi16 * 8;                     // lanes 0-15: K{0-7,16-23}; 16-31: K{8-15,24-31}
      v8bf lo = *(const v8bf*)&As[r * LSTR + kb];
      v8bf hv = *(const v8bf*)&As[r * LSTR + kb + 16];
      a[mi] = __builtin_shufflevector(lo, hv, 0,1,2,3,4,5,6,7,8,9,10,11,12,13,14,15);
    }
#pragma unroll
    for (int ni = 0; ni < 4; ++ni) {
      int r  = wn * 64 + ni * 16 + col;
      int kb = hi16 * 8;                     // lanes 0-15: K{0-7,16-23}; 16-31: K{8-15,24-31}
      v8bf lo = *(const v8bf*)&Bs[r * LSTR + kb];
      v8bf hv = *(const v8bf*)&Bs[r * LSTR + kb + 16];
      b[ni] = __builtin_shufflevector(lo, hv, 0,1,2,3,4,5,6,7,8,9,10,11,12,13,14,15);
    }

    // ---- 16 WMMAs per K step (64x64 wave tile) ----
#pragma unroll
    for (int mi = 0; mi < 4; ++mi)
#pragma unroll
      for (int ni = 0; ni < 4; ++ni)
        acc[mi][ni] = __builtin_amdgcn_wmma_f32_16x16x32_bf16(
            false, a[mi], false, b[ni], (short)0, acc[mi][ni], false, false);
  }

  // ---- epilogue: per-output-channel scale (hoisted out of dequant) + bias ----
#pragma unroll
  for (int ni = 0; ni < 4; ++ni) {
    int   n   = wgN * NT + wn * 64 + ni * 16 + col;
    float s   = scales[n];
    float bsv = bias[n];
#pragma unroll
    for (int mi = 0; mi < 4; ++mi) {
      int mbase = wgM * MT + wm * 64 + mi * 16 + hi16 * 8;
#pragma unroll
      for (int r = 0; r < 8; ++r)
        out[(size_t)(mbase + r) * OUT_F + n] = fmaf(acc[mi][ni][r], s, bsv);
    }
  }
}

extern "C" void kernel_launch(void* const* d_in, const int* in_sizes, int n_in,
                              void* d_out, int out_size, void* d_ws, size_t ws_size,
                              hipStream_t stream) {
  const float* x      = (const float*)d_in[0];
  const int*   codes  = (const int*)  d_in[1];
  const float* grid   = (const float*)d_in[2];
  const float* scales = (const float*)d_in[3];
  const float* bias   = (const float*)d_in[4];
  float*       out    = (float*)d_out;

  dim3 g(OUT_F / NT, MROWS / MT);            // 32 x 16 workgroups
  higgs_wmma_kernel<<<g, 256, 0, stream>>>(x, codes, grid, scales, bias, out);
}
